// LearnedCompressor_19490561589547
// MI455X (gfx1250) — compile-verified, matrix operand order checked by reading, not run on hardware
//
#include <hip/hip_runtime.h>
#include <math.h>

#define LDS3 __attribute__((address_space(3)))

typedef __attribute__((ext_vector_type(16))) __bf16    v16bf;
typedef __attribute__((ext_vector_type(8)))  float     v8f;
typedef __attribute__((ext_vector_type(4)))  float     f32x4;
typedef __attribute__((ext_vector_type(4)))  unsigned  u32x4;
typedef __attribute__((ext_vector_type(2)))  unsigned  u32x2;

#define DIM   128
#define HID   256
#define NQ    64
#define TT    64
#define TSEQ  4096
#define NBH   64
#define PARTIAL_FLOATS (2*NQ*DIM + 2*NQ)   // accK + accV + m + l = 16512
#define SMEM_BYTES 289024

union frag16 { v16bf v; u32x4 u[2]; unsigned short s[16]; };

__device__ __forceinline__ __bf16 f2bf(float f){
  return (__bf16)f;                       // hw f32->bf16 convert (RNE)
}
__device__ __forceinline__ unsigned short f2bfbits(float f){
  __bf16 b = (__bf16)f;
  return __builtin_bit_cast(unsigned short, b);
}

__device__ __forceinline__ float fast_rcp(float x){
#if __has_builtin(__builtin_amdgcn_rcpf)
  return __builtin_amdgcn_rcpf(x);
#else
  return 1.0f / x;
#endif
}

// Exact-form GELU with branchless rational erf (A&S 7.1.26, |err| < 1.5e-7 —
// far below the bf16 rounding applied to the result downstream).
__device__ __forceinline__ float gelu_fast(float x){
  float a = fabsf(x) * 0.70710678118654752f;     // |x|/sqrt(2)
  float t = fast_rcp(__builtin_fmaf(0.3275911f, a, 1.0f));
  float p = __builtin_fmaf(t, 1.061405429f, -1.453152027f);
  p = __builtin_fmaf(t, p, 1.421413741f);
  p = __builtin_fmaf(t, p, -0.284496736f);
  p = __builtin_fmaf(t, p, 0.254829592f);
  p = p * t;
  float e = __expf(-a * a);                      // v_mul + v_exp_f32
  float erfa = __builtin_fmaf(-p, e, 1.0f);      // erf(|x|/sqrt(2))
  float s = copysignf(erfa, x);
  return 0.5f * x * (1.0f + s);
}

// A fragment: 16x32 bf16 from row-major LDS src (ld multiple of 8, 16B aligned).
__device__ __forceinline__ v16bf load_A(const LDS3 __bf16* base, int ld, int lane){
  int m = lane & 15, kh = lane >> 4;
  const LDS3 __bf16* p = base + m * ld + kh * 8;
  frag16 f;
  f.u[0] = *(const LDS3 u32x4*)(p);
  f.u[1] = *(const LDS3 u32x4*)(p + 16);
  return f.v;
}
// B fragment from transposed storage: B[k][n] = base[n*ld + k] (contiguous in k).
__device__ __forceinline__ v16bf load_Bt(const LDS3 __bf16* base, int ld, int lane){
  int n = lane & 15, kh = lane >> 4;
  const LDS3 __bf16* p = base + n * ld + kh * 8;
  frag16 f;
  f.u[0] = *(const LDS3 u32x4*)(p);
  f.u[1] = *(const LDS3 u32x4*)(p + 16);
  return f.v;
}
// B fragment from pre-swizzled storage: tile*512 + lane*16 holds this lane's
// 16 contiguous elements.
__device__ __forceinline__ v16bf load_B_swz(const LDS3 __bf16* base, int tile, int lane){
  const LDS3 __bf16* p = base + tile * 512 + lane * 16;
  frag16 f;
  f.u[0] = *(const LDS3 u32x4*)(p);
  f.u[1] = *(const LDS3 u32x4*)(p + 8);
  return f.v;
}

__device__ __forceinline__ v8f wmma_bf16(v16bf a, v16bf b, v8f c){
  return __builtin_amdgcn_wmma_f32_16x16x32_bf16(false, a, false, b, (short)0, c,
                                                 false, false);
}

// token_proj on one [64,128] bf16 tile:
//   x @ W1 + b1 -> GELU -> @ W2 + b2 -> outs_swz (B-fragment layout)
//   and (HAS_RM) outs_rm row-major for the logits Bt path.
template <bool HAS_RM>
__device__ void token_proj_tile(const LDS3 __bf16* xs, const LDS3 __bf16* w1s,
                                const LDS3 float* b1s, const LDS3 __bf16* w2s,
                                const LDS3 float* b2s, LDS3 __bf16* hs,
                                LDS3 __bf16* outs_rm, LDS3 __bf16* outs_swz,
                                int w, int lane){
  int kh = lane >> 4, col = lane & 15;
  // mm1: [64,128] x [128,256] -> hs (bias + GELU), 64 tiles / 8 waves
#pragma unroll
  for (int i = 0; i < 8; ++i){
    int id = w * 8 + i;
    int mt = id >> 4, nt = id & 15;
    v8f c = {};
#pragma unroll
    for (int ks = 0; ks < 4; ++ks){
      v16bf a = load_A(xs + mt * 16 * DIM + ks * 32, DIM, lane);
      v16bf b = load_B_swz(w1s, ks * 16 + nt, lane);
      c = wmma_bf16(a, b, c);
    }
#pragma unroll
    for (int r = 0; r < 8; ++r){
      int row = mt * 16 + r + 8 * kh;
      float x = c[r] + b1s[nt * 16 + col];
      hs[row * HID + nt * 16 + col] = f2bf(gelu_fast(x));
    }
  }
  __syncthreads();
  // mm2: [64,256] x [256,128] -> outs, 32 tiles / 8 waves
#pragma unroll
  for (int i = 0; i < 4; ++i){
    int id = w * 4 + i;
    int mt = id >> 3, nt = id & 7;
    v8f c = {};
#pragma unroll
    for (int ks = 0; ks < 8; ++ks){
      v16bf a = load_A(hs + mt * 16 * HID + ks * 32, HID, lane);
      v16bf b = load_B_swz(w2s, ks * 8 + nt, lane);
      c = wmma_bf16(a, b, c);
    }
#pragma unroll
    for (int r = 0; r < 8; ++r){
      float val = c[r] + b2s[nt * 16 + col];
      __bf16 bv = f2bf(val);
      int t = mt * 16 + r + 8 * kh;      // token index
      if (HAS_RM) outs_rm[t * DIM + nt * 16 + col] = bv;
      // swizzled B-layout write: tile (t>>5, nt), lane (kh2, d&15), elem j2
      int t31 = t & 31;
      int kh2 = (t31 >> 3) & 1;
      int j2  = ((t31 >> 4) << 3) | (t31 & 7);
      int lane2 = (kh2 << 4) | col;
      outs_swz[((t >> 5) * 8 + nt) * 512 + lane2 * 16 + j2] = bv;
    }
  }
  __syncthreads();
}

extern __shared__ char smem_raw[];

__global__ __launch_bounds__(256, 1)
void lc_partial_kernel(const float* __restrict__ K, const float* __restrict__ V,
                       const float* __restrict__ W1, const float* __restrict__ B1,
                       const float* __restrict__ W2, const float* __restrict__ B2,
                       const float* __restrict__ Qw, float* __restrict__ ws, int Tc){
  LDS3 char* smem = (LDS3 char*)smem_raw;   // single generic->LDS cast
  LDS3 __bf16* qs   = (LDS3 __bf16*)(smem + 0);        // 64x128 row-major
  LDS3 __bf16* w1s  = (LDS3 __bf16*)(smem + 16384);    // 128x256 swizzled
  LDS3 __bf16* w2s  = (LDS3 __bf16*)(smem + 81920);    // 256x128 swizzled
  LDS3 __bf16* xsK  = (LDS3 __bf16*)(smem + 147456);   // K tile row-major
  LDS3 __bf16* xsV  = (LDS3 __bf16*)(smem + 163840);   // V tile row-major
  LDS3 __bf16* hs   = (LDS3 __bf16*)(smem + 180224);   // 64x256 row-major
  LDS3 __bf16* kpr  = (LDS3 __bf16*)(smem + 212992);   // kp row-major
  LDS3 __bf16* kpz  = (LDS3 __bf16*)(smem + 229376);   // kp swizzled
  LDS3 __bf16* vpz  = (LDS3 __bf16*)(smem + 245760);   // vp swizzled
  LDS3 __bf16* ps   = (LDS3 __bf16*)(smem + 262144);   // P 64x64 row-major
  LDS3 float*  ss   = (LDS3 float*)(smem + 270336);    // logits 64x64 f32
  LDS3 float*  b1s  = (LDS3 float*)(smem + 286720);    // 256 f32
  LDS3 float*  b2s  = (LDS3 float*)(smem + 287744);    // 128 f32
  LDS3 float*  ms   = (LDS3 float*)(smem + 288256);    // 64 f32 running max
  LDS3 float*  ls   = (LDS3 float*)(smem + 288512);    // 64 f32 running sum
  LDS3 float*  scs  = (LDS3 float*)(smem + 288768);    // 64 f32 rescale

  const int tid  = threadIdx.x;
  const int lane = tid & 31;
  const int w    = tid >> 5;
  const int kh   = lane >> 4;
  const int ncol = lane & 15;
  const int bh   = blockIdx.y;          // b*16 + h
  const int h    = bh & 15;
  const int chunk = blockIdx.x;

  const float* kb = K  + (size_t)bh * TSEQ * DIM;
  const float* vb = V  + (size_t)bh * TSEQ * DIM;
  const float* qb = Qw + (size_t)h  * NQ   * DIM;

  // ---- stage W1/W2 into swizzled B-fragment layout (one-time, L2-resident) ----
  for (int u = tid; u < 2048; u += 256){
    int tile = u >> 5, li = u & 31;
    int ks = tile >> 4, nt = tile & 15;
    int kh2 = li >> 4, n = nt * 16 + (li & 15);
    frag16 f;
#pragma unroll
    for (int j = 0; j < 16; ++j){
      int k = ks * 32 + (((j & 8) << 1) | (kh2 << 3) | (j & 7));
      f.s[j] = f2bfbits(W1[k * HID + n]);
    }
    LDS3 u32x4* dst = (LDS3 u32x4*)(w1s + tile * 512 + li * 16);
    dst[0] = f.u[0]; dst[1] = f.u[1];
  }
  for (int u = tid; u < 2048; u += 256){
    int tile = u >> 5, li = u & 31;
    int ks = tile >> 3, nt = tile & 7;
    int kh2 = li >> 4, n = nt * 16 + (li & 15);
    frag16 f;
#pragma unroll
    for (int j = 0; j < 16; ++j){
      int k = ks * 32 + (((j & 8) << 1) | (kh2 << 3) | (j & 7));
      f.s[j] = f2bfbits(W2[k * DIM + n]);
    }
    LDS3 u32x4* dst = (LDS3 u32x4*)(w2s + tile * 512 + li * 16);
    dst[0] = f.u[0]; dst[1] = f.u[1];
  }
  // q row-major (A-side)
  for (int i = tid; i < (NQ * DIM) / 4; i += 256){
    float4 f = ((const float4*)qb)[i];
    u32x2 pk;
    pk[0] = (unsigned)f2bfbits(f.x) | ((unsigned)f2bfbits(f.y) << 16);
    pk[1] = (unsigned)f2bfbits(f.z) | ((unsigned)f2bfbits(f.w) << 16);
    *(LDS3 u32x2*)(qs + 4 * i) = pk;
  }
  if (tid < HID) b1s[tid] = B1[tid];
  if (tid < DIM) b2s[tid] = B2[tid];
  if (tid < NQ){ ms[tid] = -1e30f; ls[tid] = 0.0f; }

  v8f accK[4], accV[4];
  v8f zero = {};
#pragma unroll
  for (int i = 0; i < 4; ++i){ accK[i] = zero; accV[i] = zero; }
  __syncthreads();

  const int t0 = chunk * Tc;
  for (int tt = 0; tt < Tc; tt += TT){
    // ---- stage K and V tiles together: both HBM streams in flight at once ----
    {
      const float* srcK = kb + (size_t)(t0 + tt) * DIM;
      const float* srcV = vb + (size_t)(t0 + tt) * DIM;
      for (int i = tid; i < (TT * DIM) / 4; i += 256){
        float4 fk = ((const float4*)srcK)[i];
        float4 fv = ((const float4*)srcV)[i];
        u32x2 pk, pv;
        pk[0] = (unsigned)f2bfbits(fk.x) | ((unsigned)f2bfbits(fk.y) << 16);
        pk[1] = (unsigned)f2bfbits(fk.z) | ((unsigned)f2bfbits(fk.w) << 16);
        pv[0] = (unsigned)f2bfbits(fv.x) | ((unsigned)f2bfbits(fv.y) << 16);
        pv[1] = (unsigned)f2bfbits(fv.z) | ((unsigned)f2bfbits(fv.w) << 16);
        *(LDS3 u32x2*)(xsK + 4 * i) = pk;
        *(LDS3 u32x2*)(xsV + 4 * i) = pv;
      }
    }
    __syncthreads();
    token_proj_tile<true >(xsK, w1s, b1s, w2s, b2s, hs, kpr, kpz, w, lane);
    token_proj_tile<false>(xsV, w1s, b1s, w2s, b2s, hs, kpr, vpz, w, lane);

    // ---- logits: q [64,128] x kp^T [128,64] -> ss (f32), 16 tiles / 8 waves ----
#pragma unroll
    for (int i = 0; i < 2; ++i){
      int id = w * 2 + i, mt = id >> 2, nt = id & 3;
      v8f c = {};
#pragma unroll
      for (int ks = 0; ks < 4; ++ks){
        v16bf a = load_A (qs  + mt * 16 * DIM + ks * 32, DIM, lane);
        v16bf b = load_Bt(kpr + nt * 16 * DIM + ks * 32, DIM, lane);
        c = wmma_bf16(a, b, c);
      }
#pragma unroll
      for (int r = 0; r < 8; ++r){
        int row = mt * 16 + r + 8 * kh;
        ss[row * TT + nt * 16 + ncol] = c[r];
      }
    }
    __syncthreads();

    // ---- online softmax over this tile: 4 lanes per query row ----
    {
      int row = tid >> 2, seg = tid & 3;
      const LDS3 float* srow = ss + row * TT + seg * 16;
      float lm = -1e30f;
#pragma unroll
      for (int c = 0; c < 16; ++c) lm = fmaxf(lm, srow[c]);
      lm = fmaxf(lm, __shfl_xor(lm, 1, 4));
      lm = fmaxf(lm, __shfl_xor(lm, 2, 4));
      float m_old = ms[row];
      float m_new = fmaxf(m_old, lm);
      float sum = 0.0f;
      frag16 pf;
#pragma unroll
      for (int c = 0; c < 16; ++c){
        float p = __expf(srow[c] - m_new);
        pf.s[c] = f2bfbits(p);
        sum += p;
      }
      LDS3 u32x4* dst = (LDS3 u32x4*)(ps + row * TT + seg * 16);
      dst[0] = pf.u[0]; dst[1] = pf.u[1];
      sum += __shfl_xor(sum, 1, 4);
      sum += __shfl_xor(sum, 2, 4);
      if (seg == 0){
        float sc = __expf(m_old - m_new);
        scs[row] = sc;
        ls[row]  = ls[row] * sc + sum;
        ms[row]  = m_new;
      }
    }
    __syncthreads();

    // ---- rescale accumulators and add P@kp, P@vp (32 tiles / 8 waves) ----
#pragma unroll
    for (int i = 0; i < 4; ++i){
      int id = w * 4 + i, mt = id >> 3, nt = id & 7;
      const LDS3 f32x4* sp = (const LDS3 f32x4*)(scs + mt * 16 + 8 * kh);
      f32x4 s0 = sp[0], s1 = sp[1];
      accK[i][0] *= s0[0]; accK[i][1] *= s0[1]; accK[i][2] *= s0[2]; accK[i][3] *= s0[3];
      accK[i][4] *= s1[0]; accK[i][5] *= s1[1]; accK[i][6] *= s1[2]; accK[i][7] *= s1[3];
      accV[i][0] *= s0[0]; accV[i][1] *= s0[1]; accV[i][2] *= s0[2]; accV[i][3] *= s0[3];
      accV[i][4] *= s1[0]; accV[i][5] *= s1[1]; accV[i][6] *= s1[2]; accV[i][7] *= s1[3];
#pragma unroll
      for (int ks = 0; ks < 2; ++ks){
        v16bf a  = load_A(ps + mt * 16 * TT + ks * 32, TT, lane);
        v16bf bk = load_B_swz(kpz, ks * 8 + nt, lane);
        accK[i] = wmma_bf16(a, bk, accK[i]);
        v16bf bv = load_B_swz(vpz, ks * 8 + nt, lane);
        accV[i] = wmma_bf16(a, bv, accV[i]);
      }
    }
    __syncthreads();
  }

  // ---- write per-chunk partials: accK, accV (unnormalized), m, l ----
  float* wb = ws + ((size_t)bh * gridDim.x + chunk) * PARTIAL_FLOATS;
#pragma unroll
  for (int i = 0; i < 4; ++i){
    int id = w * 4 + i, mt = id >> 3, nt = id & 7;
#pragma unroll
    for (int r = 0; r < 8; ++r){
      int row  = mt * 16 + r + 8 * kh;
      int colg = nt * 16 + ncol;
      wb[row * DIM + colg]            = accK[i][r];
      wb[NQ * DIM + row * DIM + colg] = accV[i][r];
    }
  }
  if (tid < NQ){
    wb[2 * NQ * DIM + tid]      = ms[tid];
    wb[2 * NQ * DIM + NQ + tid] = ls[tid];
  }
}

// Combine split-T partials with log-sum-exp merge, normalize, write outputs.
__global__ __launch_bounds__(128)
void lc_combine_kernel(const float* __restrict__ ws, float* __restrict__ out, int S){
  int bh = blockIdx.x >> 6;
  int qi = blockIdx.x & 63;
  int d  = threadIdx.x;                    // 0..127
  const float* base = ws + (size_t)bh * S * PARTIAL_FLOATS;
  float M = -1e30f;
  for (int s = 0; s < S; ++s)
    M = fmaxf(M, base[(size_t)s * PARTIAL_FLOATS + 2 * NQ * DIM + qi]);
  float L = 0.0f, sk = 0.0f, sv = 0.0f;
  for (int s = 0; s < S; ++s){
    const float* p = base + (size_t)s * PARTIAL_FLOATS;
    float wgt = __expf(p[2 * NQ * DIM + qi] - M);
    L  += p[2 * NQ * DIM + NQ + qi] * wgt;
    sk += p[qi * DIM + d] * wgt;
    sv += p[NQ * DIM + qi * DIM + d] * wgt;
  }
  float inv = 1.0f / L;
  size_t o = ((size_t)bh * NQ + qi) * DIM + d;
  out[o] = sk * inv;                                   // compressed_k
  out[(size_t)NBH * NQ * DIM + o] = sv * inv;          // compressed_v
}

extern "C" void kernel_launch(void* const* d_in, const int* in_sizes, int n_in,
                              void* d_out, int out_size, void* d_ws, size_t ws_size,
                              hipStream_t stream){
  (void)in_sizes; (void)n_in; (void)out_size;
  const float* K  = (const float*)d_in[0];
  const float* V  = (const float*)d_in[1];
  const float* W1 = (const float*)d_in[2];
  const float* B1 = (const float*)d_in[3];
  const float* W2 = (const float*)d_in[4];
  const float* B2 = (const float*)d_in[5];
  const float* Qw = (const float*)d_in[6];

  const size_t perChunkBytes = (size_t)NBH * PARTIAL_FLOATS * sizeof(float);
  int S = 1;
  for (int cand = 16; cand >= 1; cand >>= 1){
    if ((size_t)cand * perChunkBytes <= ws_size){ S = cand; break; }
  }
  int Tc = TSEQ / S;

  dim3 grid(S, NBH);
  lc_partial_kernel<<<grid, 256, SMEM_BYTES, stream>>>(K, V, W1, B1, W2, B2, Qw,
                                                       (float*)d_ws, Tc);
  lc_combine_kernel<<<dim3(NBH * NQ), 128, 0, stream>>>((const float*)d_ws,
                                                        (float*)d_out, S);
}
